// Attention_44624710205549
// MI455X (gfx1250) — matrix operand order repaired and checked by reading, not compiled, not run on hardware
//
#include <hip/hip_runtime.h>
#include <hip/hip_bf16.h>

// ---------------------------------------------------------------------------
// Multi-dilated local attention block for MI455X (gfx1250, wave32, WMMA).
// Heavy GEMMs (q-proj, pointwise conv, out-proj; ~79 of 84 GFLOP) run on
// V_WMMA_F32_16X16X32_BF16 with f32 accumulation; large intermediates except kv (f32) are
// bf16 in HBM (pipeline is memory-bound: ~84 GFLOP vs ~300MB @ 23.3 TB/s).
// Each wave computes a 32x64 output tile: 2 A-fragments reuse every
// B-fragment -> 1.5 b128 loads per WMMA (vs 2.5 for a 16x64 tile).
// ---------------------------------------------------------------------------

typedef __bf16 bf16;
typedef __attribute__((ext_vector_type(16))) __bf16 v16bf;
typedef __attribute__((ext_vector_type(8)))  __bf16 v8bf;
typedef __attribute__((ext_vector_type(8)))  float  v8f;

#define B_   16
#define N_   3136
#define DIM  512
#define HEADS 8
#define WS_  7
#define HW_  56
#define SD_  256
#define S_   32
#define MTOT (B_ * N_)          // 50176 rows for all big GEMMs

// ---------------------------------------------------------------------------
// 1) fp32 -> bf16 elementwise convert (x)
// ---------------------------------------------------------------------------
__global__ void convert_bf16_kernel(const float* __restrict__ src,
                                    bf16* __restrict__ dst, long long n) {
  long long i = (long long)blockIdx.x * blockDim.x + threadIdx.x;
  if (i < n) dst[i] = (bf16)src[i];
}

// ---------------------------------------------------------------------------
// 2) Pack a 512x512 weight W (row-major, out-major: W[n][k]) into the WMMA
//    B-fragment layout for D = A @ W^T:
//      B[k][n] = W[n][k];  lane holds column n, 16 contiguous K values:
//      frag element e -> K = kb*32 + g*16 + e  (g = lane>=16)
//    Packed addr: wp[ ((kb*512 + n)*2 + g)*16 + e ]  -> one 32B load/lane.
// ---------------------------------------------------------------------------
__global__ void pack_weight_kernel(const float* __restrict__ w,
                                   bf16* __restrict__ wp) {
  int idx = blockIdx.x * blockDim.x + threadIdx.x;   // 16*512*2 = 16384
  if (idx >= 16 * 512 * 2) return;
  int g  = idx & 1;
  int n  = (idx >> 1) & 511;
  int kb = idx >> 10;
  const float* src = w + (long long)n * 512 + kb * 32 + g * 16;
  bf16* dst = wp + (long long)idx * 16;
#pragma unroll
  for (int e = 0; e < 16; ++e) dst[e] = (bf16)src[e];
}

// ---------------------------------------------------------------------------
// 3) WMMA bf16 GEMM:  C(M x 512) = A(M x 512, bf16 row-major) @ Bpacked
//    MODE 0: out bf16 row-major               (Q projection)
//    MODE 1: out f32 NCHW scatter + bias      (pointwise conv -> kv)
//    MODE 2: out f32 row-major + bias         (final projection -> d_out)
//    Block = 128 threads = 4 waves; each wave owns a 32x64 output tile
//    (block covers 128 rows x 64 cols).
// ---------------------------------------------------------------------------
template <int MODE>
__global__ void gemm_wmma_kernel(const bf16* __restrict__ A,
                                 const bf16* __restrict__ Bp,
                                 const float* __restrict__ bias,
                                 void* __restrict__ out) {
  const int lane = threadIdx.x & 31;
  const int wave = threadIdx.x >> 5;       // 0..3
  const int lr   = lane & 15;
  const int g    = lane >> 4;              // lane group (ISA 16-bit layouts)
  const int m0   = blockIdx.x * 128 + wave * 32;
  const int n0   = blockIdx.y * 64;

  v8f acc[2][4];
#pragma unroll
  for (int mt = 0; mt < 2; ++mt)
#pragma unroll
    for (int t = 0; t < 4; ++t)
      acc[mt][t] = (v8f){0.f,0.f,0.f,0.f,0.f,0.f,0.f,0.f};

  const bf16* Arow0 = A + (long long)(m0 + lr) * DIM;
  const bf16* Arow1 = A + (long long)(m0 + 16 + lr) * DIM;

  for (int kb = 0; kb < 16; ++kb) {
    const int k0 = kb * 32;
    // A fragments (16x32 bf16): elem e -> K = g*16 + e
    v8bf a0lo = *(const v8bf*)(Arow0 + k0 + g * 16);
    v8bf a0hi = *(const v8bf*)(Arow0 + k0 + g * 16 + 8);
    v8bf a1lo = *(const v8bf*)(Arow1 + k0 + g * 16);
    v8bf a1hi = *(const v8bf*)(Arow1 + k0 + g * 16 + 8);
    v16bf a0 = __builtin_shufflevector(a0lo, a0hi,
        0,1,2,3,4,5,6,7,8,9,10,11,12,13,14,15);
    v16bf a1 = __builtin_shufflevector(a1lo, a1hi,
        0,1,2,3,4,5,6,7,8,9,10,11,12,13,14,15);
    if (kb + 1 < 16) {  // gfx1250 global_prefetch_b8
      __builtin_prefetch(Arow0 + k0 + 32, 0, 1);
      __builtin_prefetch(Arow1 + k0 + 32, 0, 1);
    }
#pragma unroll
    for (int t = 0; t < 4; ++t) {
      const int n = n0 + t * 16 + lr;
      const bf16* pb = Bp + ((((long long)kb * 512 + n) * 2 + g) << 4);
      v16bf b = *(const v16bf*)pb;  // 32B aligned contiguous fragment
      acc[0][t] = __builtin_amdgcn_wmma_f32_16x16x32_bf16(
          false, a0, false, b, (short)0, acc[0][t], false, false);
      acc[1][t] = __builtin_amdgcn_wmma_f32_16x16x32_bf16(
          false, a1, false, b, (short)0, acc[1][t], false, false);
    }
  }

  // D layout: lanes 0-15 -> rows +0..7 (vgpr v), lanes 16-31 -> rows +8..15
#pragma unroll
  for (int mt = 0; mt < 2; ++mt) {
#pragma unroll
    for (int t = 0; t < 4; ++t) {
      const int n = n0 + t * 16 + lr;
#pragma unroll
      for (int v = 0; v < 8; ++v) {
        const int m = m0 + mt * 16 + g * 8 + v;
        float val = acc[mt][t][v];
        if (MODE == 0) {
          ((bf16*)out)[(long long)m * DIM + n] = (bf16)val;
        } else if (MODE == 1) {
          int b = m / N_, pix = m % N_;
          ((float*)out)[((long long)b * DIM + n) * N_ + pix] = val + bias[n];
        } else {
          ((float*)out)[(long long)m * DIM + n] = val + bias[n];
        }
      }
    }
  }
}

// ---------------------------------------------------------------------------
// 4) depthwise 3x3 conv (pad 1) on x viewed as NCHW; output pixel-major bf16
//    (ready to be the A matrix of the pointwise-conv GEMM). x is L2-resident
//    (103MB < 192MB L2) so the 9x re-reads stay on-chip.
// ---------------------------------------------------------------------------
__global__ void dwconv_kv_kernel(const float* __restrict__ x,
                                 const float* __restrict__ w,
                                 const float* __restrict__ bias,
                                 bf16* __restrict__ out) {
  const int m = blockIdx.x;                  // b*3136 + pix
  const int b = m / N_, pix = m % N_;
  const int y = pix / HW_, xx = pix % HW_;
  const int c = blockIdx.y * 256 + threadIdx.x;
  float acc = bias[c];
#pragma unroll
  for (int ky = 0; ky < 3; ++ky) {
    int yy = y + ky - 1;
    if (yy < 0 || yy >= HW_) continue;
#pragma unroll
    for (int kx = 0; kx < 3; ++kx) {
      int xn = xx + kx - 1;
      if (xn < 0 || xn >= HW_) continue;
      acc += w[c * 9 + ky * 3 + kx] *
             x[((long long)b * N_ + yy * HW_ + xn) * DIM + c];
    }
  }
  out[(long long)m * DIM + c] = (bf16)acc;
}

// ---------------------------------------------------------------------------
// 5) strided "pool": replicates the reference's (..,L2*S)->(..,S,L2) mean,
//    which interleaves location l and channel s:
//    pooled[br,b,h,p,j] = 1/L2 * sum_t kv[b, br*256+h*32+s, y, x]
//      with f = j*L2+t, l=f/32, s=f%32, (ly,lx)=l/L,l%L, (ky,kx)=p/7,p%7,
//      y = ly*stride+ky*dil, x = lx*stride+kx*dil.
// ---------------------------------------------------------------------------
__global__ void pool_kernel(const float* __restrict__ kv,
                            float* __restrict__ pooled) {
  int idx = blockIdx.x * 256 + threadIdx.x;
  const int TOT = 2 * B_ * HEADS * 49 * 32;
  if (idx >= TOT) return;
  int j  = idx & 31;
  int p  = (idx >> 5) % 49;
  int h  = (idx / (49 * 32)) % HEADS;
  int b  = (idx / (49 * 32 * HEADS)) % B_;
  int br = idx / (49 * 32 * HEADS * B_);
  const int dil = 1 << br;
  const int stride = dil * (WS_ - 1) + 1;
  const int L  = br ? 4 : 8;
  const int L2 = L * L;
  const int ky = p / 7, kx = p % 7;
  float s = 0.f;
  for (int t = 0; t < L2; ++t) {
    int f  = j * L2 + t;
    int l  = f >> 5;
    int ss = f & 31;
    int ly = l / L, lx = l % L;
    int yy = ly * stride + ky * dil;
    int xn = lx * stride + kx * dil;
    int c  = br * SD_ + h * S_ + ss;
    s += kv[((long long)b * DIM + c) * N_ + yy * HW_ + xn];
  }
  pooled[idx] = s / (float)L2;
}

// ---------------------------------------------------------------------------
// 6) per-(branch,b,head): kvp = pooled @ fc^T @ single^T; split k,v;
//    v += dwconv3x3(v as 7x7 image) + bias. All tiny; LDS-resident.
// ---------------------------------------------------------------------------
__global__ void kv_head_kernel(const float* __restrict__ pooled,
                               const float* __restrict__ fc0,
                               const float* __restrict__ fc1,
                               const float* __restrict__ sw0,
                               const float* __restrict__ sw1,
                               const float* __restrict__ lw0,
                               const float* __restrict__ lb0,
                               const float* __restrict__ lw1,
                               const float* __restrict__ lb1,
                               float* __restrict__ kbuf,
                               float* __restrict__ vbuf) {
  const int blk = blockIdx.x;            // ((br*16)+b)*8 + h
  const int h  = blk & 7;
  const int br = blk >> 7;
  const float* fc = br ? fc1 : fc0;
  const float* sw = br ? sw1 : sw0;
  const float* lw = br ? lw1 : lw0;
  const float* lb = br ? lb1 : lb0;

  __shared__ float sp[49 * 32], st[49 * 32], skvp[49 * 64];
  __shared__ float sfc[32 * 32], ssw[64 * 32];
  const int tid = threadIdx.x;           // 256
  const float* pbase = pooled + (long long)blk * 49 * 32;
  for (int i = tid; i < 49 * 32; i += 256) sp[i]  = pbase[i];
  for (int i = tid; i < 32 * 32; i += 256) sfc[i] = fc[i];
  for (int i = tid; i < 64 * 32; i += 256) ssw[i] = sw[i];
  __syncthreads();
  for (int i = tid; i < 49 * 32; i += 256) {
    int p = i >> 5, s2 = i & 31;
    float a = 0.f;
    for (int s = 0; s < 32; ++s) a += sp[p * 32 + s] * sfc[s2 * 32 + s];
    st[i] = a;
  }
  __syncthreads();
  for (int i = tid; i < 49 * 64; i += 256) {
    int p = i / 64, o = i % 64;
    float a = 0.f;
    for (int s2 = 0; s2 < 32; ++s2) a += st[p * 32 + s2] * ssw[o * 32 + s2];
    skvp[i] = a;
  }
  __syncthreads();
  float* kdst = kbuf + (long long)blk * 49 * 32;
  float* vdst = vbuf + (long long)blk * 49 * 32;
  for (int i = tid; i < 49 * 32; i += 256) {
    int p = i >> 5, s = i & 31;
    kdst[i] = skvp[p * 64 + s];
    int c = h * S_ + s;
    float acc = lb[c];
    int py = p / 7, px = p % 7;
#pragma unroll
    for (int ky = 0; ky < 3; ++ky) {
      int yy = py + ky - 1;
      if (yy < 0 || yy >= 7) continue;
#pragma unroll
      for (int kx = 0; kx < 3; ++kx) {
        int xn = px + kx - 1;
        if (xn < 0 || xn >= 7) continue;
        acc += lw[c * 9 + ky * 3 + kx] * skvp[(yy * 7 + xn) * 64 + 32 + s];
      }
    }
    vdst[i] = skvp[p * 64 + 32 + s] + acc;
  }
}

// ---------------------------------------------------------------------------
// 7) fused softmax-attention: 49 keys/values in LDS, one query row / thread.
//    Writes concat output as bf16 (A matrix of the final WMMA GEMM).
// ---------------------------------------------------------------------------
__global__ void attn_kernel(const bf16* __restrict__ Q,
                            const float* __restrict__ kbuf,
                            const float* __restrict__ vbuf,
                            bf16* __restrict__ out) {
  const int b  = blockIdx.z;
  const int h  = blockIdx.y & 7;
  const int br = blockIdx.y >> 3;
  const int blk = ((br * B_) + b) * HEADS + h;
  __shared__ float sk[49 * 32], sv[49 * 32];
  for (int i = threadIdx.x; i < 49 * 32; i += blockDim.x) {
    sk[i] = kbuf[(long long)blk * 1568 + i];
    sv[i] = vbuf[(long long)blk * 1568 + i];
  }
  __syncthreads();
  const int n = blockIdx.x * blockDim.x + threadIdx.x;
  if (n >= N_) return;

  // 64B-aligned q row: four b128 loads via v8bf
  const bf16* qp = Q + ((long long)b * N_ + n) * DIM + h * 64 + br * S_;
  v8bf q0 = *(const v8bf*)(qp);
  v8bf q1 = *(const v8bf*)(qp + 8);
  v8bf q2 = *(const v8bf*)(qp + 16);
  v8bf q3 = *(const v8bf*)(qp + 24);
  float q[32];
#pragma unroll
  for (int s = 0; s < 8; ++s) {
    q[s]      = (float)q0[s];
    q[s + 8]  = (float)q1[s];
    q[s + 16] = (float)q2[s];
    q[s + 24] = (float)q3[s];
  }

  float sc[49];
  float mx = -1e30f;
  for (int j = 0; j < 49; ++j) {
    float a = 0.f;
#pragma unroll
    for (int s = 0; s < 32; ++s) a += q[s] * sk[j * 32 + s];
    a *= 0.17677669529663687f;   // 32^-0.5
    sc[j] = a;
    mx = fmaxf(mx, a);
  }
  float den = 0.f;
  for (int j = 0; j < 49; ++j) {
    float e = __expf(sc[j] - mx);
    sc[j] = e;
    den += e;
  }
  const float inv = 1.f / den;
  bf16* op = out + ((long long)b * N_ + n) * DIM + br * SD_ + h * S_;
#pragma unroll
  for (int s = 0; s < 32; ++s) {
    float a = 0.f;
    for (int j = 0; j < 49; ++j) a += sc[j] * sv[j * 32 + s];
    op[s] = (bf16)(a * inv);
  }
}

// ---------------------------------------------------------------------------
// host launcher
// ---------------------------------------------------------------------------
extern "C" void kernel_launch(void* const* d_in, const int* in_sizes, int n_in,
                              void* d_out, int out_size, void* d_ws,
                              size_t ws_size, hipStream_t stream) {
  (void)in_sizes; (void)n_in; (void)out_size; (void)ws_size;
  const float* x      = (const float*)d_in[0];
  // d_in[1]=H, d_in[2]=W (constants)
  const float* q_w    = (const float*)d_in[3];
  const float* dw_w   = (const float*)d_in[4];
  const float* dw_b   = (const float*)d_in[5];
  const float* pw_w   = (const float*)d_in[6];
  const float* pw_b   = (const float*)d_in[7];
  const float* fc_w0  = (const float*)d_in[8];
  const float* fc_w1  = (const float*)d_in[9];
  const float* sw0    = (const float*)d_in[10];
  const float* sw1    = (const float*)d_in[11];
  const float* lw0    = (const float*)d_in[12];
  const float* lb0    = (const float*)d_in[13];
  const float* lw1    = (const float*)d_in[14];
  const float* lb1    = (const float*)d_in[15];
  const float* proj_w = (const float*)d_in[16];
  const float* proj_b = (const float*)d_in[17];
  float* outp = (float*)d_out;

  // workspace layout (bytes)
  const long long NB_BF = (long long)MTOT * DIM * 2;   // 51,380,224
  const long long NB_F  = (long long)MTOT * DIM * 4;   // 102,760,448
  char* ws = (char*)d_ws;
  bf16*  xbf    = (bf16*)(ws);
  bf16*  qbf    = (bf16*)(ws + NB_BF);
  bf16*  dwbf   = (bf16*)(ws + 2 * NB_BF);
  float* kvf    = (float*)(ws + 3 * NB_BF);
  bf16*  aobf   = (bf16*)(ws + 3 * NB_BF + NB_F);
  bf16*  wq     = (bf16*)(ws + 4 * NB_BF + NB_F);
  bf16*  wpw    = (bf16*)(ws + 4 * NB_BF + NB_F + 524288);
  bf16*  wproj  = (bf16*)(ws + 4 * NB_BF + NB_F + 2 * 524288);
  float* pooled = (float*)(ws + 4 * NB_BF + NB_F + 3 * 524288);
  float* kbuf   = (float*)(ws + 4 * NB_BF + NB_F + 3 * 524288 + 1605632);
  float* vbuf   = (float*)(ws + 4 * NB_BF + NB_F + 3 * 524288 + 2 * 1605632);

  const long long NELEM = (long long)MTOT * DIM;

  // 1) x -> bf16
  convert_bf16_kernel<<<dim3((unsigned)((NELEM + 255) / 256)), 256, 0, stream>>>(
      x, xbf, NELEM);

  // 2) pack weights into WMMA B-fragment layout
  pack_weight_kernel<<<64, 256, 0, stream>>>(q_w, wq);
  pack_weight_kernel<<<64, 256, 0, stream>>>(pw_w, wpw);
  pack_weight_kernel<<<64, 256, 0, stream>>>(proj_w, wproj);

  // 3) Q = x @ q_w^T  (bf16 out)
  gemm_wmma_kernel<0><<<dim3(MTOT / 128, DIM / 64), 128, 0, stream>>>(
      xbf, wq, nullptr, (void*)qbf);

  // 4) depthwise conv -> pixel-major bf16
  dwconv_kv_kernel<<<dim3(MTOT, 2), 256, 0, stream>>>(x, dw_w, dw_b, dwbf);

  // 5) pointwise conv GEMM -> kv (f32, NCHW) + bias
  gemm_wmma_kernel<1><<<dim3(MTOT / 128, DIM / 64), 128, 0, stream>>>(
      dwbf, wpw, pw_b, (void*)kvf);

  // 6) strided pool (replicates reference's interleaved mean)
  pool_kernel<<<dim3((2 * B_ * HEADS * 49 * 32 + 255) / 256), 256, 0, stream>>>(
      kvf, pooled);

  // 7) per-head fc/single/local-dwconv -> k, v
  kv_head_kernel<<<dim3(2 * B_ * HEADS), 256, 0, stream>>>(
      pooled, fc_w0, fc_w1, sw0, sw1, lw0, lb0, lw1, lb1, kbuf, vbuf);

  // 8) fused softmax attention -> bf16 concat output
  attn_kernel<<<dim3((N_ + 255) / 256, 2 * HEADS, B_), 256, 0, stream>>>(
      qbf, kbuf, vbuf, aobf);

  // 9) out = attn_cat @ proj_w^T + proj_b  (f32 to d_out)
  gemm_wmma_kernel<2><<<dim3(MTOT / 128, DIM / 64), 128, 0, stream>>>(
      aobf, wproj, proj_b, (void*)outp);
}
